// TransLayer_9706626089325
// MI455X (gfx1250) — matrix operand order repaired and checked by reading, not compiled, not run on hardware
//
#include <hip/hip_runtime.h>
#include <cstdint>

typedef __attribute__((ext_vector_type(16))) _Float16 v16h;
typedef __attribute__((ext_vector_type(8)))  _Float16 v8h;
typedef __attribute__((ext_vector_type(8)))  float    v8f;

#define B_    4
#define N_    4096
#define D_    512
#define H_    8
#define DH_   64
#define M_    256
#define KW_   33
#define BH_   (B_ * H_)
#define BN_   (B_ * N_)

// Issue one per-lane async 16B memory->LDS transfer (CDNA5 TDM-lite path).
// ldsOff: byte offset of this lane's LDS destination; gaddr: lane's global VA.
__device__ __forceinline__ void async_copy16(uint32_t ldsOff, uint64_t gaddr)
{
  asm volatile("global_load_async_to_lds_b128 %0, %1, off"
               :: "v"(ldsOff), "v"(gaddr) : "memory");
}
__device__ __forceinline__ void async_copy16_off16(uint32_t ldsOff, uint64_t gaddr)
{
  asm volatile("global_load_async_to_lds_b128 %0, %1, off offset:16"
               :: "v"(ldsOff), "v"(gaddr) : "memory");
}
__device__ __forceinline__ void wait_async0()
{
  asm volatile("s_wait_asynccnt 0x0" ::: "memory");
}
__device__ __forceinline__ uint32_t lds_off(const void* p)
{
  return (uint32_t)(uintptr_t)p;   // low 32 bits of generic shared VA = LDS offset
}

// ---------------------------------------------------------------------------
// Batched WMMA GEMM: C[Md x Nd] = scale * A[Md x Kd] * B  (+ fused epilogues)
//   BT=false: B is [Kd x Nd] row-major (NN)
//   BT=true : B is [Nd x Kd] row-major (NT, i.e. A * B^T)
// Double-buffered LDS staging; A (and B in NT mode) staged with
// global_load_async_to_lds_b128, overlapping the WMMA compute of the
// previous K-step. NN-mode B needs an n-major transpose, staged synchronously.
// Epilogues (any may be null):
//   outH   : f16,  outH   = (h)(scale*acc)
//   outSub : f16,  outSub = (h)(dval*(row==col) - scale*acc)   [pinv terms]
//   outF   : f32,  outF   = scale*acc (+ bias[col]) (+ resid[row*Nd+col])
// All of Md%64==0, Nd%64==0, Kd%32==0 hold for every call in this file.
// ---------------------------------------------------------------------------
template<bool BT>
__global__ __launch_bounds__(128)
void gemm_f16(const _Float16* __restrict__ A, const _Float16* __restrict__ B,
              long long sA, long long sB, long long sC,
              int Md, int Nd, int Kd, float scale,
              _Float16* __restrict__ outH,
              _Float16* __restrict__ outSub, float dval,
              float* __restrict__ outF,
              const float* __restrict__ bias,
              const float* __restrict__ resid)
{
  // As stored with k contiguous (slot s = k) so an A fragment is a contiguous 32B LDS read:
  //   k[0..31]->s[0..31]; half-wave g reads k[16g..16g+15] (same order as Bs)
  // Bs stored n-major so a B fragment is a contiguous 32B LDS read.
  __shared__ __attribute__((aligned(32))) _Float16 As[2][64][32];
  __shared__ __attribute__((aligned(32))) _Float16 Bs[2][64][32];

  const int batch = blockIdx.z;
  A += (long long)batch * sA;
  B += (long long)batch * sB;
  const long long cbase = (long long)batch * sC;

  const int t    = threadIdx.x;
  const int wave = t >> 5;
  const int lane = t & 31;
  const int g    = lane >> 4;   // half-wave group (ISA fragment layout)
  const int lr   = lane & 15;
  const int m0   = blockIdx.y * 64;
  const int n0   = blockIdx.x * 64;
  const int wm   = (wave >> 1) * 32;
  const int wn   = (wave & 1) * 32;

  // per-thread staging coordinates
  const int rowA = t >> 1;               // 0..63
  const int kkA  = (t & 1) * 16;         // half index within 32-k tile
  const int slA  = kkA;                  // contiguous slot base: 0 or 16
  const int nB   = t >> 1;               // NT: 0..63
  const int kkB  = (t & 1) * 16;         // NT
  const int rkB  = t >> 2;               // NN: k row 0..31
  const int nnB  = (t & 3) * 16;         // NN

  auto stage = [&](int buf, int k0) {
    { // A tile, async, contiguous k per row
      const _Float16* src = A + (long long)(m0 + rowA) * Kd + (k0 + kkA);
      const uint64_t ga = (uint64_t)(uintptr_t)src;
      async_copy16(lds_off(&As[buf][rowA][slA]), ga);
      async_copy16_off16(lds_off(&As[buf][rowA][slA + 8]), ga);
    }
    if (BT) { // B tile, async, contiguous k per n
      const _Float16* src = B + (long long)(n0 + nB) * Kd + (k0 + kkB);
      const uint64_t gb = (uint64_t)(uintptr_t)src;
      async_copy16(lds_off(&Bs[buf][nB][kkB]), gb);
      async_copy16_off16(lds_off(&Bs[buf][nB][kkB + 8]), gb);
    } else {  // B tile, synchronous transpose scatter into Bs[n][k]
      const _Float16* src = B + (long long)(k0 + rkB) * Nd + (n0 + nnB);
#pragma unroll
      for (int e = 0; e < 16; ++e) Bs[buf][nnB + e][rkB] = src[e];
    }
  };

  v8f acc[2][2];
#pragma unroll
  for (int i = 0; i < 2; ++i)
#pragma unroll
    for (int j = 0; j < 2; ++j)
      acc[i][j] = (v8f){0.f,0.f,0.f,0.f,0.f,0.f,0.f,0.f};

  int cur = 0;
  stage(0, 0);
  wait_async0();
  __syncthreads();

  for (int k0 = 0; k0 < Kd; k0 += 32) {
    const int  nxt      = cur ^ 1;
    const bool has_next = (k0 + 32) < Kd;
    if (has_next) stage(nxt, k0 + 32);   // fills other buffer during compute

    v16h af[2], bf[2];
#pragma unroll
    for (int tm = 0; tm < 2; ++tm)
      af[tm] = *(const v16h*)&As[cur][wm + tm * 16 + lr][g * 16];
#pragma unroll
    for (int tn = 0; tn < 2; ++tn)
      bf[tn] = *(const v16h*)&Bs[cur][wn + tn * 16 + lr][g * 16];

#pragma unroll
    for (int tm = 0; tm < 2; ++tm)
#pragma unroll
      for (int tn = 0; tn < 2; ++tn)
        acc[tm][tn] = __builtin_amdgcn_wmma_f32_16x16x32_f16(
            false, af[tm], false, bf[tn], (short)0, acc[tm][tn], false, false);

    if (has_next) {
      wait_async0();      // next-stage async transfers complete
      __syncthreads();    // visible to all waves; cur readers done before overwrite
      cur = nxt;
    }
  }

  // ---- epilogue (C/D layout: VGPR r -> row r+8g, lane -> col lane&15)
#pragma unroll
  for (int tm = 0; tm < 2; ++tm)
#pragma unroll
    for (int tn = 0; tn < 2; ++tn) {
      const int gcol = n0 + wn + tn * 16 + lr;
#pragma unroll
      for (int r = 0; r < 8; ++r) {
        const int grow = m0 + wm + tm * 16 + r + 8 * g;
        const float va = acc[tm][tn][r] * scale;
        const long long idx = cbase + (long long)grow * Nd + gcol;
        if (outH)   outH[idx]   = (_Float16)va;
        if (outSub) outSub[idx] = (_Float16)(((grow == gcol) ? dval : 0.f) - va);
        if (outF) {
          float o = va;
          if (bias)  o += bias[gcol];
          if (resid) o += resid[(long long)grow * Nd + gcol];
          outF[idx] = o;
        }
      }
    }
}

// ---------------------------------------------------------------------------
__global__ __launch_bounds__(256)
void k_layernorm(const float* __restrict__ x, const float* __restrict__ w,
                 const float* __restrict__ bch, _Float16* __restrict__ out)
{
  const int row = blockIdx.x;
  const float* xr = x + (long long)row * D_;
  __shared__ float red[256];
  float s = 0.f;
  for (int c = threadIdx.x; c < D_; c += 256) s += xr[c];
  red[threadIdx.x] = s; __syncthreads();
  for (int o = 128; o > 0; o >>= 1) { if (threadIdx.x < o) red[threadIdx.x] += red[threadIdx.x + o]; __syncthreads(); }
  const float mu = red[0] * (1.f / D_);
  __syncthreads();
  float v = 0.f;
  for (int c = threadIdx.x; c < D_; c += 256) { float d = xr[c] - mu; v += d * d; }
  red[threadIdx.x] = v; __syncthreads();
  for (int o = 128; o > 0; o >>= 1) { if (threadIdx.x < o) red[threadIdx.x] += red[threadIdx.x + o]; __syncthreads(); }
  const float rstd = rsqrtf(red[0] * (1.f / D_) + 1e-5f);
  _Float16* orow = out + (long long)row * D_;
  for (int c = threadIdx.x; c < D_; c += 256)
    orow[c] = (_Float16)((xr[c] - mu) * rstd * w[c] + bch[c]);
}

__global__ __launch_bounds__(256)
void k_softmax(const float* __restrict__ in, float* __restrict__ outF,
               _Float16* __restrict__ outH, int cols)
{
  const long long row = blockIdx.x;
  const float* r = in + row * cols;
  __shared__ float red[256];
  float mx = -3.4e38f;
  for (int c = threadIdx.x; c < cols; c += 256) mx = fmaxf(mx, r[c]);
  red[threadIdx.x] = mx; __syncthreads();
  for (int o = 128; o > 0; o >>= 1) { if (threadIdx.x < o) red[threadIdx.x] = fmaxf(red[threadIdx.x], red[threadIdx.x + o]); __syncthreads(); }
  mx = red[0]; __syncthreads();
  float s = 0.f;
  for (int c = threadIdx.x; c < cols; c += 256) s += __expf(r[c] - mx);
  red[threadIdx.x] = s; __syncthreads();
  for (int o = 128; o > 0; o >>= 1) { if (threadIdx.x < o) red[threadIdx.x] += red[threadIdx.x + o]; __syncthreads(); }
  const float inv = 1.f / red[0];
  for (int c = threadIdx.x; c < cols; c += 256) {
    const float e = __expf(r[c] - mx) * inv;
    if (outF) outF[row * cols + c] = e;
    if (outH) outH[row * cols + c] = (_Float16)e;
  }
}

__global__ void k_cvt16(const float* __restrict__ s, _Float16* __restrict__ d, long long n)
{
  const long long i = (long long)blockIdx.x * blockDim.x + threadIdx.x;
  if (i < n) d[i] = (_Float16)s[i];
}

// qkv [BN,1536] f16 -> q,k,v [B,H,N,DH] f16; q pre-scaled by DH^-0.5
__global__ void k_split_qkv(const _Float16* __restrict__ qkv,
                            _Float16* __restrict__ q, _Float16* __restrict__ k,
                            _Float16* __restrict__ v, float qscale)
{
  const long long i = (long long)blockIdx.x * blockDim.x + threadIdx.x;
  if (i >= (long long)BH_ * N_ * DH_) return;
  const int d = (int)(i % DH_);
  long long tt = i / DH_;
  const int n = (int)(tt % N_); tt /= N_;
  const int h = (int)(tt % H_);
  const int b = (int)(tt / H_);
  const _Float16* base = qkv + ((long long)b * N_ + n) * (3 * D_) + h * DH_ + d;
  q[i] = (_Float16)((float)base[0] * qscale);
  k[i] = base[D_];
  v[i] = base[2 * D_];
}

// landmark means: [B,H,N,DH] -> [B,H,M,DH], mean over N_/M_ = 16 tokens
__global__ void k_landmark(const _Float16* __restrict__ src, _Float16* __restrict__ dst)
{
  const long long i = (long long)blockIdx.x * blockDim.x + threadIdx.x;
  if (i >= (long long)BH_ * M_ * DH_) return;
  const int d = (int)(i % DH_);
  long long tt = i / DH_;
  const int m = (int)(tt % M_);
  const long long bh = tt / M_;
  const _Float16* p = src + ((long long)bh * N_ + m * (N_ / M_)) * DH_ + d;
  float s = 0.f;
#pragma unroll
  for (int j = 0; j < N_ / M_; ++j) s += (float)p[(long long)j * DH_];
  dst[i] = (_Float16)(s * (float)M_ / (float)N_);
}

__global__ void k_zero2(float* s) { if (threadIdx.x < 2) s[threadIdx.x] = 0.f; }

// per-(b,h): max abs row-sum (sum over -1) and col-sum (sum over -2), global max
__global__ __launch_bounds__(256)
void k_a2_maxsums(const float* __restrict__ a2, float* __restrict__ scal)
{
  const int bh = blockIdx.x;
  const float* Am = a2 + (long long)bh * M_ * M_;
  const int tt = threadIdx.x;
  float rs = 0.f, cs = 0.f;
  for (int j = 0; j < M_; ++j) {
    rs += fabsf(Am[tt * M_ + j]);
    cs += fabsf(Am[j * M_ + tt]);
  }
  __shared__ float rmax[256], cmax[256];
  rmax[tt] = rs; cmax[tt] = cs; __syncthreads();
  for (int o = 128; o > 0; o >>= 1) {
    if (tt < o) { rmax[tt] = fmaxf(rmax[tt], rmax[tt + o]); cmax[tt] = fmaxf(cmax[tt], cmax[tt + o]); }
    __syncthreads();
  }
  if (tt == 0) {  // non-negative floats: uint compare == float compare
    atomicMax((unsigned int*)&scal[0], __float_as_uint(rmax[0]));
    atomicMax((unsigned int*)&scal[1], __float_as_uint(cmax[0]));
  }
}

// z0 = a2^T / (max_rowsum * max_colsum)
__global__ void k_init_z(const float* __restrict__ a2, const float* __restrict__ scal,
                         _Float16* __restrict__ z)
{
  const long long i = (long long)blockIdx.x * blockDim.x + threadIdx.x;
  if (i >= (long long)BH_ * M_ * M_) return;
  const int j = (int)(i % M_);
  long long tt = i / M_;
  const int r = (int)(tt % M_);
  const long long bh = tt / M_;
  const float inv = 1.f / (scal[0] * scal[1]);
  z[i] = (_Float16)(a2[(bh * M_ + j) * M_ + r] * inv);
}

// depthwise conv over sequence (K=33, pad 16), attn += conv(v)
__global__ void k_conv_add(const _Float16* __restrict__ v, const float* __restrict__ wconv,
                           float* __restrict__ attn)
{
  const long long i = (long long)blockIdx.x * blockDim.x + threadIdx.x;
  if (i >= (long long)BH_ * N_ * DH_) return;
  const int d = (int)(i % DH_);
  long long tt = i / DH_;
  const int n = (int)(tt % N_);
  const long long bh = tt / N_;
  const int h = (int)(bh % H_);
  const _Float16* vb = v + bh * ((long long)N_ * DH_) + d;
  const float* wk = wconv + h * KW_;
  float s = 0.f;
#pragma unroll
  for (int kk = 0; kk < KW_; ++kk) {
    const int np = n + kk - KW_ / 2;
    if (np >= 0 && np < N_) s += (float)vb[(long long)np * DH_] * wk[kk];
  }
  attn[i] += s;
}

// attn [B,H,N,DH] f32 -> merged heads [B,N,H*DH] f16
__global__ void k_merge(const float* __restrict__ attn, _Float16* __restrict__ mh)
{
  const long long i = (long long)blockIdx.x * blockDim.x + threadIdx.x;
  if (i >= (long long)BN_ * D_) return;
  const int d = (int)(i % DH_);
  long long tt = i / DH_;
  const int h = (int)(tt % H_); tt /= H_;
  const int n = (int)(tt % N_);
  const int b = (int)(tt / N_);
  mh[i] = (_Float16)attn[((((long long)b * H_ + h) * N_) + n) * DH_ + d];
}

// ---------------------------------------------------------------------------
static void launch_gemm(bool bt, const _Float16* A, const _Float16* B,
                        long long sA, long long sB, long long sC,
                        int Md, int Nd, int Kd, float scale,
                        _Float16* oh, _Float16* osub, float dval,
                        float* of, const float* bias, const float* resid,
                        int batches, hipStream_t st)
{
  dim3 grid(Nd / 64, Md / 64, batches), block(128);
  if (bt)
    gemm_f16<true ><<<grid, block, 0, st>>>(A, B, sA, sB, sC, Md, Nd, Kd, scale, oh, osub, dval, of, bias, resid);
  else
    gemm_f16<false><<<grid, block, 0, st>>>(A, B, sA, sB, sC, Md, Nd, Kd, scale, oh, osub, dval, of, bias, resid);
}

static inline size_t alup(size_t x) { return (x + 255) & ~(size_t)255; }

extern "C" void kernel_launch(void* const* d_in, const int* in_sizes, int n_in,
                              void* d_out, int out_size, void* d_ws, size_t ws_size,
                              hipStream_t stream)
{
  const float* x     = (const float*)d_in[0];
  const float* lnw   = (const float*)d_in[1];
  const float* lnb   = (const float*)d_in[2];
  const float* wqkv  = (const float*)d_in[3];
  const float* wout  = (const float*)d_in[4];
  const float* bout  = (const float*)d_in[5];
  const float* wconv = (const float*)d_in[6];
  float* out = (float*)d_out;

  // ---- workspace carve-up
  char* p = (char*)d_ws;
  auto take = [&](size_t bytes) -> char* { char* r = p; p += alup(bytes); return r; };
  const long long QS   = (long long)BH_ * N_ * DH_;   // q/k/v elements
  const long long LS   = (long long)BH_ * M_ * DH_;   // landmark elements
  const long long SIMS = (long long)BH_ * N_ * M_;    // sim1/sim3 elements
  const long long MM   = (long long)BH_ * M_ * M_;    // 256x256 batched

  _Float16* xnh   = (_Float16*)take((size_t)BN_ * D_ * 2);
  _Float16* wqkvh = (_Float16*)take((size_t)D_ * 3 * D_ * 2);
  _Float16* wouth = (_Float16*)take((size_t)D_ * D_ * 2);
  _Float16* qkvh  = (_Float16*)take((size_t)BN_ * 3 * D_ * 2);
  _Float16* qh    = (_Float16*)take((size_t)QS * 2);
  _Float16* kh    = (_Float16*)take((size_t)QS * 2);
  _Float16* vh    = (_Float16*)take((size_t)QS * 2);
  _Float16* qlh   = (_Float16*)take((size_t)LS * 2);
  _Float16* klh   = (_Float16*)take((size_t)LS * 2);
  float*    simf  = (float*)take((size_t)SIMS * 4);          // reused later
  _Float16* a1h   = (_Float16*)take((size_t)SIMS * 2);
  _Float16* a3h   = (_Float16*)take((size_t)SIMS * 2);
  float*    a2f   = (float*)take((size_t)MM * 4);
  _Float16* a2h   = (_Float16*)take((size_t)MM * 2);
  _Float16* z0    = (_Float16*)take((size_t)MM * 2);
  _Float16* z1    = (_Float16*)take((size_t)MM * 2);
  _Float16* xzh   = (_Float16*)take((size_t)MM * 2);
  _Float16* t1h   = (_Float16*)take((size_t)MM * 2);
  _Float16* t2h   = (_Float16*)take((size_t)MM * 2);
  _Float16* t3h   = (_Float16*)take((size_t)MM * 2);
  _Float16* a3vh  = (_Float16*)take((size_t)LS * 2);
  float*    scal  = (float*)take(256);
  // aliases into simf (free after a3 softmax): a1@z result, attn, merged heads
  _Float16* th    = (_Float16*)simf;
  float*    attnf = (float*)((char*)simf + alup((size_t)SIMS * 2));
  _Float16* mh    = (_Float16*)((char*)attnf + alup((size_t)QS * 4));

  const int TPB = 256;
  auto nblk = [&](long long n) { return (unsigned)((n + TPB - 1) / TPB); };

  // ---- weights to f16, LayerNorm
  k_cvt16<<<nblk((long long)D_ * 3 * D_), TPB, 0, stream>>>(wqkv, wqkvh, (long long)D_ * 3 * D_);
  k_cvt16<<<nblk((long long)D_ * D_), TPB, 0, stream>>>(wout, wouth, (long long)D_ * D_);
  k_layernorm<<<BN_, TPB, 0, stream>>>(x, lnw, lnb, xnh);

  // ---- QKV projection: [16384x512]x[512x1536] -> f16
  launch_gemm(false, xnh, wqkvh, 0, 0, 0, BN_, 3 * D_, D_, 1.f,
              qkvh, nullptr, 0.f, nullptr, nullptr, nullptr, 1, stream);
  k_split_qkv<<<nblk(QS), TPB, 0, stream>>>(qkvh, qh, kh, vh, 0.125f);  // DH^-0.5
  k_landmark<<<nblk(LS), TPB, 0, stream>>>(qh, qlh);
  k_landmark<<<nblk(LS), TPB, 0, stream>>>(kh, klh);

  // ---- sim1 = q @ kl^T -> softmax -> a1 (f16)
  launch_gemm(true, qh, klh, QS / BH_, LS / BH_, SIMS / BH_, N_, M_, DH_, 1.f,
              nullptr, nullptr, 0.f, simf, nullptr, nullptr, BH_, stream);
  k_softmax<<<(unsigned)(BH_ * N_), TPB, 0, stream>>>(simf, nullptr, a1h, M_);

  // ---- sim2 = ql @ kl^T -> softmax -> a2 (f32 + f16)
  launch_gemm(true, qlh, klh, LS / BH_, LS / BH_, MM / BH_, M_, M_, DH_, 1.f,
              nullptr, nullptr, 0.f, a2f, nullptr, nullptr, BH_, stream);
  k_softmax<<<(unsigned)(BH_ * M_), TPB, 0, stream>>>(a2f, a2f, a2h, M_);

  // ---- sim3 = ql @ k^T -> softmax -> a3 (f16)
  launch_gemm(true, qlh, kh, LS / BH_, QS / BH_, SIMS / BH_, M_, N_, DH_, 1.f,
              nullptr, nullptr, 0.f, simf, nullptr, nullptr, BH_, stream);
  k_softmax<<<(unsigned)(BH_ * M_), TPB, 0, stream>>>(simf, nullptr, a3h, N_);

  // ---- Moore-Penrose pinv (Newton-Schulz, 6 iters, batched 256^3 WMMA GEMMs)
  k_zero2<<<1, 32, 0, stream>>>(scal);
  k_a2_maxsums<<<BH_, TPB, 0, stream>>>(a2f, scal);
  k_init_z<<<nblk(MM), TPB, 0, stream>>>(a2f, scal, z0);
  const long long sM = MM / BH_;
  _Float16* zc = z0; _Float16* zn = z1;
  for (int it = 0; it < 6; ++it) {
    // xz = a2 @ z ; t1 = 7I - xz
    launch_gemm(false, a2h, zc, sM, sM, sM, M_, M_, M_, 1.f,
                xzh, t1h, 7.f, nullptr, nullptr, nullptr, BH_, stream);
    // t2 = 15I - xz @ t1
    launch_gemm(false, xzh, t1h, sM, sM, sM, M_, M_, M_, 1.f,
                nullptr, t2h, 15.f, nullptr, nullptr, nullptr, BH_, stream);
    // t3 = 13I - xz @ t2
    launch_gemm(false, xzh, t2h, sM, sM, sM, M_, M_, M_, 1.f,
                nullptr, t3h, 13.f, nullptr, nullptr, nullptr, BH_, stream);
    // z = 0.25 * z @ t3
    launch_gemm(false, zc, t3h, sM, sM, sM, M_, M_, M_, 0.25f,
                zn, nullptr, 0.f, nullptr, nullptr, nullptr, BH_, stream);
    _Float16* tmp = zc; zc = zn; zn = tmp;
  }

  // ---- a3v = a3 @ v  [256x4096]x[4096x64]
  launch_gemm(false, a3h, vh, SIMS / BH_, QS / BH_, LS / BH_, M_, DH_, N_, 1.f,
              a3vh, nullptr, 0.f, nullptr, nullptr, nullptr, BH_, stream);
  // ---- th = a1 @ z   [4096x256]x[256x256]
  launch_gemm(false, a1h, zc, SIMS / BH_, sM, SIMS / BH_, N_, M_, M_, 1.f,
              th, nullptr, 0.f, nullptr, nullptr, nullptr, BH_, stream);
  // ---- attn = th @ a3v  [4096x256]x[256x64] -> f32
  launch_gemm(false, th, a3vh, SIMS / BH_, LS / BH_, QS / BH_, N_, DH_, M_, 1.f,
              nullptr, nullptr, 0.f, attnf, nullptr, nullptr, BH_, stream);

  // ---- depthwise residual conv over sequence, merge heads
  k_conv_add<<<nblk(QS), TPB, 0, stream>>>(vh, wconv, attnf);
  k_merge<<<nblk((long long)BN_ * D_), TPB, 0, stream>>>(attnf, mh);

  // ---- out = mh @ w_out + b_out + x   (fused epilogue, f32 to d_out)
  launch_gemm(false, mh, wouth, 0, 0, 0, BN_, D_, D_, 1.f,
              nullptr, nullptr, 0.f, out, bout, x, 1, stream);
}
